// ImprovedTransformerLayer_2164663517860
// MI455X (gfx1250) — matrix operand order repaired and checked by reading, not compiled, not run on hardware
//
#include <hip/hip_runtime.h>

// ---------------- CDNA5 types ----------------
typedef __attribute__((ext_vector_type(16))) __bf16 bf16x16;
typedef __attribute__((ext_vector_type(8)))  float  v8f;

constexpr int kS   = 2048;
constexpr int kD   = 1024;
constexpr int kH   = 16;
constexpr int kHD  = 64;
constexpr int kDFF = 4096;

__device__ __forceinline__ unsigned short f2bf(float f) {
  unsigned u = __builtin_bit_cast(unsigned, f);
  u += 0x7FFFu + ((u >> 16) & 1u);          // round-to-nearest-even
  return (unsigned short)(u >> 16);
}

union Frag { bf16x16 v; uint4 q[2]; };

// ---------------- CDNA5 async global->LDS copy (ASYNCcnt path) ----------------
#if defined(__gfx1250__) && __has_builtin(__builtin_amdgcn_global_load_async_to_lds_b128)
#define HAS_ASYNC_LDS 1
#else
#define HAS_ASYNC_LDS 0
#endif

typedef int v4i __attribute__((vector_size(16)));
typedef __attribute__((address_space(1))) v4i* gptr_v4i;   // global v4i*
typedef __attribute__((address_space(3))) v4i* lptr_v4i;   // LDS v4i*

// copy 16 bytes global -> LDS; async on CDNA5, register round-trip otherwise
__device__ __forceinline__ void copy16_g2l(const unsigned short* g, unsigned short* l) {
#if HAS_ASYNC_LDS
  __builtin_amdgcn_global_load_async_to_lds_b128(
      (gptr_v4i)(uintptr_t)g, (lptr_v4i)(uintptr_t)l, 0, 0);
#else
  *(uint4*)l = *(const uint4*)g;
#endif
}

__device__ __forceinline__ void async_wait0() {
#if HAS_ASYNC_LDS
#if __has_builtin(__builtin_amdgcn_s_wait_asynccnt)
  __builtin_amdgcn_s_wait_asynccnt(0);
#else
  asm volatile("s_wait_asynccnt 0" ::: "memory");
#endif
#endif
}

// ---------------- fp32 -> bf16 conversion ----------------
__global__ void cvt_f32_bf16(const float* __restrict__ in,
                             unsigned short* __restrict__ out, int n) {
  int i = blockIdx.x * blockDim.x + threadIdx.x;
  int stride = gridDim.x * blockDim.x;
  for (; i < n; i += stride) out[i] = f2bf(in[i]);
}

// ---------------- LayerNorm (one row / block of 256) ----------------
__global__ void layernorm_bf16(const float* __restrict__ x,
                               const float* __restrict__ g,
                               const float* __restrict__ b,
                               unsigned short* __restrict__ out) {
  int row = blockIdx.x;
  int tid = threadIdx.x;                      // 256 threads, 4 floats each
  float4 v = ((const float4*)(x + (size_t)row * kD))[tid];
  float s  = v.x + v.y + v.z + v.w;
  float s2 = v.x * v.x + v.y * v.y + v.z * v.z + v.w * v.w;
#pragma unroll
  for (int off = 16; off >= 1; off >>= 1) {
    s  += __shfl_xor(s,  off, 32);
    s2 += __shfl_xor(s2, off, 32);
  }
  __shared__ float red0[8], red1[8];
  int wave = tid >> 5, lane = tid & 31;
  if (lane == 0) { red0[wave] = s; red1[wave] = s2; }
  __syncthreads();
  float ts = 0.f, ts2 = 0.f;
#pragma unroll
  for (int w = 0; w < 8; ++w) { ts += red0[w]; ts2 += red1[w]; }
  float mean = ts * (1.0f / kD);
  float var  = ts2 * (1.0f / kD) - mean * mean;
  float rstd = rsqrtf(var + 1e-5f);
  float4 gv = ((const float4*)g)[tid];
  float4 bv = ((const float4*)b)[tid];
  unsigned short* o = out + (size_t)row * kD + tid * 4;
  o[0] = f2bf((v.x - mean) * rstd * gv.x + bv.x);
  o[1] = f2bf((v.y - mean) * rstd * gv.y + bv.y);
  o[2] = f2bf((v.z - mean) * rstd * gv.z + bv.z);
  o[3] = f2bf((v.w - mean) * rstd * gv.w + bv.w);
}

// ---------------- bf16 tiled GEMM:  C = A[M,K] @ W[N,K]^T ----------------
// Double-buffered LDS pipeline. On CDNA5 the next K-step streams straight
// into the alternate LDS stage via GLOBAL_LOAD_ASYNC_TO_LDS_B128 while WMMAs
// consume the current stage; s_wait_asynccnt + barrier closes the stage.
// MODE 0: store bf16          MODE 1: +bias, GELU, store bf16
// MODE 2: +bias +resid, f32   MODE 3: +resid, f32
template <int MODE>
__global__ __launch_bounds__(256, 1)
void gemm_bf16(const unsigned short* __restrict__ A,
               const unsigned short* __restrict__ W,
               int M, int N, int K,
               const float* __restrict__ bias,
               const float* __restrict__ resid,
               unsigned short* __restrict__ out_bf,
               float* __restrict__ out_f) {
  __shared__ unsigned short As[2][128][40];   // padded: 80B row stride
  __shared__ unsigned short Bs[2][128][40];
  const int m0 = blockIdx.y * 128, n0 = blockIdx.x * 128;
  const int tid  = threadIdx.x;               // 256 threads, 8 waves
  const int wave = tid >> 5, lane = tid & 31;
  const int wm = wave >> 2, wn = wave & 3;    // 2 x 4 wave grid
  const int c = lane & 15, h = lane >> 4;

  v8f acc[4][2];
#pragma unroll
  for (int a = 0; a < 4; ++a)
#pragma unroll
    for (int b2 = 0; b2 < 2; ++b2) acc[a][b2] = {};

  const int lrow = tid >> 1;
  const int lk   = (tid & 1) * 16;
  const unsigned short* ag = A + (size_t)(m0 + lrow) * K + lk;
  const unsigned short* bg = W + (size_t)(n0 + lrow) * K + lk;

  // prologue: stage 0
  copy16_g2l(ag,     &As[0][lrow][lk]);
  copy16_g2l(ag + 8, &As[0][lrow][lk + 8]);
  copy16_g2l(bg,     &Bs[0][lrow][lk]);
  copy16_g2l(bg + 8, &Bs[0][lrow][lk + 8]);
  async_wait0();
  __syncthreads();

  const int nkt = K >> 5;
  for (int kt = 0; kt < nkt; ++kt) {
    const int cur = kt & 1, nxt = cur ^ 1;
    const bool more = (kt + 1) < nkt;
#if !HAS_ASYNC_LDS
    uint4 na0, na1, nb0, nb1;
#endif
    if (more) {                               // stream next K-step
      ag += 32; bg += 32;
#if HAS_ASYNC_LDS
      copy16_g2l(ag,     &As[nxt][lrow][lk]);
      copy16_g2l(ag + 8, &As[nxt][lrow][lk + 8]);
      copy16_g2l(bg,     &Bs[nxt][lrow][lk]);
      copy16_g2l(bg + 8, &Bs[nxt][lrow][lk + 8]);
#else
      __builtin_prefetch(ag + 64, 0, 1);
      __builtin_prefetch(bg + 64, 0, 1);
      na0 = *(const uint4*)(ag);
      na1 = *(const uint4*)(ag + 8);
      nb0 = *(const uint4*)(bg);
      nb1 = *(const uint4*)(bg + 8);
#endif
    }

    Frag fa[4], fb[2];
#pragma unroll
    for (int mi = 0; mi < 4; ++mi) {          // 16-bit A 16x32 lane layout
      const unsigned short* p = &As[cur][wm * 64 + mi * 16 + c][h * 8];
      fa[mi].q[0] = *(const uint4*)p;
      fa[mi].q[1] = *(const uint4*)(p + 16);
    }
#pragma unroll
    for (int ni = 0; ni < 2; ++ni) {          // B 32x16: column = W row
      const unsigned short* p = &Bs[cur][wn * 32 + ni * 16 + c][h * 8];
      fb[ni].q[0] = *(const uint4*)p;
      fb[ni].q[1] = *(const uint4*)(p + 16);
    }
#pragma unroll
    for (int mi = 0; mi < 4; ++mi)
#pragma unroll
      for (int ni = 0; ni < 2; ++ni)
        acc[mi][ni] = __builtin_amdgcn_wmma_f32_16x16x32_bf16(
            false, fa[mi].v, false, fb[ni].v, (short)0, acc[mi][ni], false, false);

#if !HAS_ASYNC_LDS
    if (more) {                               // fill the other stage
      *(uint4*)&As[nxt][lrow][lk]     = na0;
      *(uint4*)&As[nxt][lrow][lk + 8] = na1;
      *(uint4*)&Bs[nxt][lrow][lk]     = nb0;
      *(uint4*)&Bs[nxt][lrow][lk + 8] = nb1;
    }
#endif
    async_wait0();
    __syncthreads();
  }

#pragma unroll
  for (int mi = 0; mi < 4; ++mi)
#pragma unroll
    for (int ni = 0; ni < 2; ++ni)
#pragma unroll
      for (int i = 0; i < 8; ++i) {
        int row = m0 + wm * 64 + mi * 16 + i + 8 * h;
        int col = n0 + wn * 32 + ni * 16 + c;
        float v = acc[mi][ni][i];
        if (MODE == 1 || MODE == 2) v += bias[col];
        if (MODE == 1) v = 0.5f * v * (1.0f + erff(v * 0.70710678118f));
        if (MODE == 2 || MODE == 3) v += resid[(size_t)row * N + col];
        if (MODE == 0 || MODE == 1) out_bf[(size_t)row * N + col] = f2bf(v);
        else                        out_f[(size_t)row * N + col]  = v;
      }
}

// ---------------- Flash attention: 64-query block x 1 head per WG ----------------
__global__ __launch_bounds__(128, 1)
void attn_kernel(const unsigned short* __restrict__ Q,
                 const unsigned short* __restrict__ Kb,
                 const unsigned short* __restrict__ V,
                 const float* __restrict__ slopes,
                 unsigned short* __restrict__ O) {
  __shared__ unsigned short qT[64][72];
  __shared__ unsigned short kT[64][72];
  __shared__ unsigned short vT[64][72];        // transposed: [hd][key]
  __shared__ unsigned short pT[4][16][72];     // per-wave P relayout buffer
  const int qb = blockIdx.x, head = blockIdx.y;
  const int q0 = qb * 64, col0 = head * 64;
  const int tid = threadIdx.x;                 // 128 threads, 4 waves
  const int wave = tid >> 5, lane = tid & 31;
  const int c = lane & 15, h = lane >> 4;
  const float slope = slopes[head];

  for (int idx = tid; idx < 64 * 8; idx += 128) {
    int r = idx >> 3, seg = idx & 7;
    copy16_g2l(&Q[(size_t)(q0 + r) * kD + col0 + seg * 8], &qT[r][seg * 8]);
  }
  async_wait0();
  __syncthreads();

  v8f o[4];
  float m[8], l[8];
#pragma unroll
  for (int t = 0; t < 4; ++t) o[t] = {};
#pragma unroll
  for (int i = 0; i < 8; ++i) { m[i] = -1e30f; l[i] = 0.0f; }
  const int rowb = q0 + wave * 16;
  float rowf[8];
#pragma unroll
  for (int i = 0; i < 8; ++i) rowf[i] = (float)(rowb + i + 8 * h);

  for (int k0 = 0; k0 <= q0; k0 += 64) {       // causal: only lower blocks
    // K tile: async stream into LDS; V tile transposed via registers (overlaps)
    for (int idx = tid; idx < 64 * 8; idx += 128) {
      int r = idx >> 3, seg = idx & 7;
      copy16_g2l(&Kb[(size_t)(k0 + r) * kD + col0 + seg * 8], &kT[r][seg * 8]);
    }
    for (int idx = tid; idx < 64 * 8; idx += 128) {
      int r = idx >> 3, seg = idx & 7;
      uint4 vv = *(const uint4*)&V[(size_t)(k0 + r) * kD + col0 + seg * 8];
      const unsigned short* hw = (const unsigned short*)&vv;
#pragma unroll
      for (int e = 0; e < 8; ++e) vT[seg * 8 + e][r] = hw[e];
    }
    async_wait0();
    __syncthreads();

    // scores = Q @ K^T : reload Q frags from LDS each block (no spills)
    Frag fq[2];
#pragma unroll
    for (int kk = 0; kk < 2; ++kk) {
      const unsigned short* p = &qT[wave * 16 + c][kk * 32 + h * 8];
      fq[kk].q[0] = *(const uint4*)p;
      fq[kk].q[1] = *(const uint4*)(p + 16);
    }
    v8f s[4];
#pragma unroll
    for (int t = 0; t < 4; ++t) s[t] = {};
#pragma unroll
    for (int nt = 0; nt < 4; ++nt)
#pragma unroll
      for (int kk = 0; kk < 2; ++kk) {
        Frag fk;
        const unsigned short* p = &kT[nt * 16 + c][kk * 32 + h * 8];
        fk.q[0] = *(const uint4*)p;
        fk.q[1] = *(const uint4*)(p + 16);
        s[nt] = __builtin_amdgcn_wmma_f32_16x16x32_bf16(
            false, fq[kk].v, false, fk.v, (short)0, s[nt], false, false);
      }

    // scale + ALiBi (mask only needed on the diagonal block)
#pragma unroll
    for (int nt = 0; nt < 4; ++nt) {
      float colf = (float)(k0 + nt * 16 + c);
#pragma unroll
      for (int i = 0; i < 8; ++i)
        s[nt][i] = s[nt][i] * 0.125f - slope * (colf - rowf[i]);
    }
    if (k0 == q0) {
#pragma unroll
      for (int nt = 0; nt < 4; ++nt) {
        int col = k0 + nt * 16 + c;
#pragma unroll
        for (int i = 0; i < 8; ++i)
          if (col > rowb + i + 8 * h) s[nt][i] = -1e30f;
      }
    }

    float alpha[8], psum[8];
#pragma unroll
    for (int i = 0; i < 8; ++i) {
      float mx = fmaxf(fmaxf(s[0][i], s[1][i]), fmaxf(s[2][i], s[3][i]));
#pragma unroll
      for (int off = 1; off < 16; off <<= 1)   // stays inside 16-lane half
        mx = fmaxf(mx, __shfl_xor(mx, off, 32));
      float mn = fmaxf(m[i], mx);
      alpha[i] = __expf(m[i] - mn);
      m[i] = mn;
      psum[i] = 0.0f;
    }
#pragma unroll
    for (int nt = 0; nt < 4; ++nt)
#pragma unroll
      for (int i = 0; i < 8; ++i) {
        float p = __expf(s[nt][i] - m[i]);
        psum[i] += p;
        pT[wave][i + 8 * h][nt * 16 + c] = f2bf(p);
      }
#pragma unroll
    for (int i = 0; i < 8; ++i) {
      float ps = psum[i];
#pragma unroll
      for (int off = 1; off < 16; off <<= 1) ps += __shfl_xor(ps, off, 32);
      l[i] = l[i] * alpha[i] + ps;
#pragma unroll
      for (int t = 0; t < 4; ++t) o[t][i] *= alpha[i];
    }

    Frag fp[2];
#pragma unroll
    for (int kk = 0; kk < 2; ++kk) {
      const unsigned short* p = &pT[wave][c][kk * 32 + h * 8];
      fp[kk].q[0] = *(const uint4*)p;
      fp[kk].q[1] = *(const uint4*)(p + 16);
    }
#pragma unroll
    for (int t = 0; t < 4; ++t)
#pragma unroll
      for (int kk = 0; kk < 2; ++kk) {
        Frag fv;
        const unsigned short* p = &vT[t * 16 + c][kk * 32 + h * 8];
        fv.q[0] = *(const uint4*)p;
        fv.q[1] = *(const uint4*)(p + 16);
        o[t] = __builtin_amdgcn_wmma_f32_16x16x32_bf16(
            false, fp[kk].v, false, fv.v, (short)0, o[t], false, false);
      }
    __syncthreads();
  }

#pragma unroll
  for (int t = 0; t < 4; ++t)
#pragma unroll
    for (int i = 0; i < 8; ++i) {
      int row = rowb + i + 8 * h;
      O[(size_t)row * kD + col0 + t * 16 + c] = f2bf(o[t][i] / l[i]);
    }
}

// ---------------- host side ----------------
extern "C" void kernel_launch(void* const* d_in, const int* in_sizes, int n_in,
                              void* d_out, int out_size, void* d_ws, size_t ws_size,
                              hipStream_t stream) {
  (void)in_sizes; (void)n_in; (void)out_size; (void)ws_size;
  const float* x    = (const float*)d_in[0];
  const float* slp  = (const float*)d_in[1];
  const float* wq   = (const float*)d_in[2];
  const float* wk   = (const float*)d_in[3];
  const float* wv   = (const float*)d_in[4];
  const float* wo   = (const float*)d_in[5];
  const float* ff1w = (const float*)d_in[6];
  const float* ff1b = (const float*)d_in[7];
  const float* ff2w = (const float*)d_in[8];
  const float* ff2b = (const float*)d_in[9];
  const float* ln1g = (const float*)d_in[10];
  const float* ln1b = (const float*)d_in[11];
  const float* ln2g = (const float*)d_in[12];
  const float* ln2b = (const float*)d_in[13];
  float* out = (float*)d_out;

  char* ws = (char*)d_ws;
  size_t off = 0;
  auto alloc = [&](size_t bytes) {
    void* p = ws + off;
    off += (bytes + 255) & ~(size_t)255;
    return p;
  };
  unsigned short* wq_b  = (unsigned short*)alloc((size_t)kD * kD * 2);
  unsigned short* wk_b  = (unsigned short*)alloc((size_t)kD * kD * 2);
  unsigned short* wv_b  = (unsigned short*)alloc((size_t)kD * kD * 2);
  unsigned short* wo_b  = (unsigned short*)alloc((size_t)kD * kD * 2);
  unsigned short* f1w_b = (unsigned short*)alloc((size_t)kDFF * kD * 2);
  unsigned short* f2w_b = (unsigned short*)alloc((size_t)kD * kDFF * 2);
  unsigned short* h1_b  = (unsigned short*)alloc((size_t)kS * kD * 2);
  unsigned short* q_b   = (unsigned short*)alloc((size_t)kS * kD * 2);
  unsigned short* k_b   = (unsigned short*)alloc((size_t)kS * kD * 2);
  unsigned short* v_b   = (unsigned short*)alloc((size_t)kS * kD * 2);
  unsigned short* at_b  = (unsigned short*)alloc((size_t)kS * kD * 2);
  float*          x2    = (float*)alloc((size_t)kS * kD * 4);
  unsigned short* h2_b  = (unsigned short*)alloc((size_t)kS * kD * 2);
  unsigned short* g_b   = (unsigned short*)alloc((size_t)kS * kDFF * 2);

  cvt_f32_bf16<<<1024, 256, 0, stream>>>(wq,   wq_b,  kD * kD);
  cvt_f32_bf16<<<1024, 256, 0, stream>>>(wk,   wk_b,  kD * kD);
  cvt_f32_bf16<<<1024, 256, 0, stream>>>(wv,   wv_b,  kD * kD);
  cvt_f32_bf16<<<1024, 256, 0, stream>>>(wo,   wo_b,  kD * kD);
  cvt_f32_bf16<<<2048, 256, 0, stream>>>(ff1w, f1w_b, kDFF * kD);
  cvt_f32_bf16<<<2048, 256, 0, stream>>>(ff2w, f2w_b, kD * kDFF);

  layernorm_bf16<<<kS, 256, 0, stream>>>(x, ln1g, ln1b, h1_b);

  dim3 gQKV(kD / 128, kS / 128);
  gemm_bf16<0><<<gQKV, 256, 0, stream>>>(h1_b, wq_b, kS, kD, kD, nullptr, nullptr, q_b, nullptr);
  gemm_bf16<0><<<gQKV, 256, 0, stream>>>(h1_b, wk_b, kS, kD, kD, nullptr, nullptr, k_b, nullptr);
  gemm_bf16<0><<<gQKV, 256, 0, stream>>>(h1_b, wv_b, kS, kD, kD, nullptr, nullptr, v_b, nullptr);

  attn_kernel<<<dim3(kS / 64, kH), 128, 0, stream>>>(q_b, k_b, v_b, slp, at_b);

  gemm_bf16<3><<<gQKV, 256, 0, stream>>>(at_b, wo_b, kS, kD, kD, nullptr, x, nullptr, x2);

  layernorm_bf16<<<kS, 256, 0, stream>>>(x2, ln2g, ln2b, h2_b);

  gemm_bf16<1><<<dim3(kDFF / 128, kS / 128), 256, 0, stream>>>(
      h2_b, f1w_b, kS, kDFF, kD, ff1b, nullptr, g_b, nullptr);
  gemm_bf16<2><<<gQKV, 256, 0, stream>>>(
      g_b, f2w_b, kS, kD, kDFF, ff2b, x2, nullptr, out);
}
